// BasicMultiUpdateBlock_7301444403534
// MI455X (gfx1250) — compile-verified, matrix operand order checked by reading, not run on hardware
//
#include <hip/hip_runtime.h>
#include <hip/hip_bf16.h>

// ---------------- CDNA5 (gfx1250) implicit-GEMM conv via WMMA ----------------
// wave32; WMMA f32 = A(16x32 f16 weights) x B(32x16 f16 pixels) + C(16x16 f32)
// Activations/intermediates: NHWC f16 (channels innermost, padded to mult of 32)

typedef __attribute__((ext_vector_type(16))) _Float16 v16h;
typedef __attribute__((ext_vector_type(8)))  _Float16 v8h;
typedef __attribute__((ext_vector_type(8)))  float    v8f;
typedef int v4i_vs __attribute__((vector_size(16)));   // matches builtin param

#define NB_  8
#define H_   160
#define W_   320
#define HW_  (H_ * W_)
#define NPTX (NB_ * H_ * (W_ / 64))   // pixel tiles: 64 px along W, one row each

#define ACT_NONE 0
#define ACT_RELU 1
#define ACT_SIGM 2
#define ACT_TANH 3

// LDS tile: 64 pixels x 32 channels (halves); row stride 40 halves (80B,
// 16B-aligned) to spread banks. Double buffered for async pipelining.
#define LDS_STRIDE 40

// ---- CDNA5 async global->LDS copy (guarded; sync fallback otherwise) ----
#if defined(__gfx1250__) && __has_builtin(__builtin_amdgcn_global_load_async_to_lds_b128)
#define USE_ASYNC_LDS 1
#else
#define USE_ASYNC_LDS 0
#endif

__device__ __forceinline__ void copy16_g2l(const _Float16* g, _Float16* l) {
#if USE_ASYNC_LDS
    __builtin_amdgcn_global_load_async_to_lds_b128(
        (__attribute__((address_space(1))) v4i_vs*)g,
        (__attribute__((address_space(3))) v4i_vs*)l, 0, 0);
#else
    *(v8h*)l = *(const v8h*)g;
#endif
}

__device__ __forceinline__ void wait_async_le1() {
#if USE_ASYNC_LDS
#if __has_builtin(__builtin_amdgcn_s_wait_asynccnt)
    __builtin_amdgcn_s_wait_asynccnt((short)1);
#else
    asm volatile("s_wait_asynccnt 1" ::: "memory");
#endif
#endif
}
__device__ __forceinline__ void wait_async_le0() {
#if USE_ASYNC_LDS
#if __has_builtin(__builtin_amdgcn_s_wait_asynccnt)
    __builtin_amdgcn_s_wait_asynccnt((short)0);
#else
    asm volatile("s_wait_asynccnt 0" ::: "memory");
#endif
#endif
}

// Pack OIHW f32 weights -> f16 A-fragments: wp[((t*nck + ck)*Coutp + o)*32 + k],
// c = ck*32 + k; zero padded for o >= Cout or c >= Cin.  (nck*32 == padded Cin)
__global__ void pack_weights(const float* __restrict__ w, _Float16* __restrict__ wp,
                             int Cout, int Cin, int KH, int KW) {
    int Coutp = (Cout + 15) & ~15;
    int nck   = (Cin + 31) >> 5;
    size_t total = (size_t)KH * KW * nck * Coutp * 32;
    for (size_t idx = (size_t)blockIdx.x * 256 + threadIdx.x; idx < total;
         idx += (size_t)gridDim.x * 256) {
        int    k = (int)(idx & 31);
        size_t r = idx >> 5;
        int    o = (int)(r % Coutp); r /= Coutp;
        int   ck = (int)(r % nck);   r /= nck;
        int    t = (int)r;
        int   kh = t / KW, kw = t % KW;
        int    c = (ck << 5) + k;
        float  v = 0.f;
        if (o < Cout && c < Cin)
            v = w[(((size_t)o * Cin + c) * KH + kh) * KW + kw];
        wp[idx] = (_Float16)v;
    }
}

// Generic conv. Input = concat of up to 3 NHWC f16 segments with padded
// channel counts C0p/C1p/C2p (multiples of 32; 32-chunks never straddle).
// Output: NHWC f16 (outh) or NCHW f32 (outf).
__global__ __launch_bounds__(256)
void conv_wmma(const _Float16* __restrict__ s0, int C0p,
               const _Float16* __restrict__ s1, int C1p,
               const _Float16* __restrict__ s2, int C2p,
               const _Float16* __restrict__ wp,
               const float* __restrict__ bias,
               const _Float16* __restrict__ zpad,   // >=16B of zeros
               _Float16* __restrict__ outh, int outHCs, int outHCoff,
               float* __restrict__ outf, int outFCs,
               int Cout, int KH, int KW, int pad, int dil, int act, float scale) {
    const int Cinp  = C0p + C1p + C2p;
    const int nck   = Cinp >> 5;
    const int Coutp = (Cout + 15) & ~15;
    const int mtiles = Coutp >> 4;

    __shared__ _Float16 lds[2][64 * LDS_STRIDE];

    const int tid  = threadIdx.x;
    const int lane = tid & 31;
    const int wave = tid >> 5;

    // pixel tile -> (b, h, w0)
    int pt    = blockIdx.x;
    int wtile = pt % (W_ / 64);
    int bh    = pt / (W_ / 64);
    int h     = bh % H_;
    int b     = bh / H_;
    int w0    = wtile * 64;

    int  mtile  = wave + 8 * blockIdx.y;
    bool active = (mtile < mtiles);
    int  o0     = mtile * 16;

    v8f acc[4] = {};

    // staging mapping: thread -> pixel sn (0..63), 8-channel group q8 (0,8,16,24)
    const int sn = tid & 63;
    const int q8 = (tid >> 6) << 3;

    const int hi    = lane >> 4;      // half-wave select per WMMA 16-bit layout
    const int l15   = lane & 15;
    const int koff1 = hi ? 8 : 0;     // K {0..7,16..23} vs {8..15,24..31}

    const int total_it = KH * KW * nck;

    auto stage = [&](int it, int bufsel) {
        int t  = it / nck, ck = it - t * nck;
        int kh = t / KW,   kw = t - kh * KW;
        int ih = h + dil * kh - pad;
        int iw = w0 + sn + dil * kw - pad;
        const _Float16* gp;
        if (ih < 0 || ih >= H_ || iw < 0 || iw >= W_) {
            gp = zpad;                               // zero-filled scratch line
        } else {
            int c0 = (ck << 5) + q8;                 // padded-channel index
            const _Float16* seg; int cb, Cs;
            if (c0 < C0p)            { seg = s0; cb = c0;              Cs = C0p; }
            else if (c0 < C0p + C1p) { seg = s1; cb = c0 - C0p;        Cs = C1p; }
            else                     { seg = s2; cb = c0 - C0p - C1p;  Cs = C2p; }
            gp = seg + ((size_t)b * HW_ + (size_t)ih * W_ + iw) * Cs + cb;
        }
        copy16_g2l(gp, &lds[bufsel][sn * LDS_STRIDE + q8]);
    };

    stage(0, 0);
    for (int it = 0; it < total_it; ++it) {
        int cur = it & 1;
        bool more = (it + 1 < total_it);
        if (more) stage(it + 1, cur ^ 1);            // prefetch next tile
        if (more) wait_async_le1(); else wait_async_le0();
        __syncthreads();                             // tile[cur] visible to all

        if (active) {
            int t  = it / nck, ck = it - t * nck;
            // A fragment (packed weights), documented 16-bit A layout
            const _Float16* ap =
                wp + ((size_t)(t * nck + ck) * Coutp + o0 + l15) * 32;
            __builtin_prefetch(ap + (size_t)Coutp * 32, 0, 1);
            v8h alo = *(const v8h*)(ap + koff1);
            v8h ahi = *(const v8h*)(ap + koff1 + 16);
            v16h A;
#pragma unroll
            for (int i = 0; i < 8; ++i) { A[i] = alo[i]; A[i + 8] = ahi[i]; }

#pragma unroll
            for (int sub = 0; sub < 4; ++sub) {
                const _Float16* bp = &lds[cur][(sub * 16 + l15) * LDS_STRIDE + koff1];
                v8h blo = *(const v8h*)bp;
                v8h bhi = *(const v8h*)(bp + 16);
                v16h Bm;
#pragma unroll
                for (int i = 0; i < 8; ++i) { Bm[i] = blo[i]; Bm[i + 8] = bhi[i]; }
                acc[sub] = __builtin_amdgcn_wmma_f32_16x16x32_f16(
                    false, A, false, Bm, (short)0, acc[sub], false, false);
            }
        }
        __syncthreads();                             // before cur is restaged
    }

    if (active) {
        size_t pixrow = (size_t)b * HW_ + (size_t)h * W_;
#pragma unroll
        for (int sub = 0; sub < 4; ++sub) {
            int px = w0 + sub * 16 + l15;
#pragma unroll
            for (int r = 0; r < 8; ++r) {
                int o = o0 + r + hi * 8;             // C/D layout: lane>=16 -> M+8
                if (o < Cout) {
                    float v = acc[sub][r] + (bias ? bias[o] : 0.f);
                    v *= scale;
                    if (act == ACT_RELU)      v = fmaxf(v, 0.f);
                    else if (act == ACT_SIGM) v = 1.f / (1.f + __expf(-v));
                    else if (act == ACT_TANH) v = tanhf(v);
                    if (outh)
                        outh[(pixrow + px) * outHCs + outHCoff + o] = (_Float16)v;
                    else
                        outf[((size_t)b * outFCs + o) * HW_ +
                             (size_t)h * W_ + px] = v;
                }
            }
        }
    }
}

// ---------------- layout / elementwise helpers ----------------
__global__ void zero_h(_Float16* __restrict__ p, size_t n) {
    for (size_t i = (size_t)blockIdx.x * 256 + threadIdx.x; i < n;
         i += (size_t)gridDim.x * 256) p[i] = (_Float16)0.f;
}

// NCHW f32 -> NHWC f16 with channel padding (c >= C -> 0)
__global__ void to_nhwc_h(const float* __restrict__ src, _Float16* __restrict__ dst,
                          int C, int Cpad) {
    size_t total = (size_t)NB_ * HW_ * Cpad;
    for (size_t i = (size_t)blockIdx.x * 256 + threadIdx.x; i < total;
         i += (size_t)gridDim.x * 256) {
        int    c = (int)(i % Cpad);
        size_t p = (i / Cpad) % HW_;
        size_t b = i / ((size_t)Cpad * HW_);
        float  v = (c < C) ? src[((size_t)b * C + c) * HW_ + p] : 0.f;
        dst[i] = (_Float16)v;
    }
}

__global__ void flow_to_motion(const float* __restrict__ flow,
                               _Float16* __restrict__ motion) {
    size_t n = (size_t)NB_ * HW_;
    size_t i = (size_t)blockIdx.x * 256 + threadIdx.x;
    if (i < n) motion[i * 64 + 63] = (_Float16)flow[i];  // motion ch 63 = flow
}

__global__ void ew_mul_h(const _Float16* __restrict__ a, const _Float16* __restrict__ b,
                         _Float16* __restrict__ o, size_t n) {
    for (size_t i = (size_t)blockIdx.x * 256 + threadIdx.x; i < n;
         i += (size_t)gridDim.x * 256)
        o[i] = (_Float16)((float)a[i] * (float)b[i]);
}

// h = (1-z)*net0 + z*q ; write NCHW f32 (d_out) and NHWC f16 (for heads)
__global__ void gru_combine(const _Float16* __restrict__ z, const _Float16* __restrict__ q,
                            const _Float16* __restrict__ net0h,
                            float* __restrict__ hf, _Float16* __restrict__ hh) {
    size_t total = (size_t)NB_ * HW_ * 128;
    for (size_t i = (size_t)blockIdx.x * 256 + threadIdx.x; i < total;
         i += (size_t)gridDim.x * 256) {
        int    c = (int)(i % 128);
        size_t p = (i / 128) % HW_;
        size_t b = i / ((size_t)128 * HW_);
        float zv = (float)z[i], qv = (float)q[i], nv = (float)net0h[i];
        float hv = (1.f - zv) * nv + zv * qv;
        hh[i] = (_Float16)hv;
        hf[((size_t)b * 128 + c) * HW_ + p] = hv;
    }
}

// ---------------- host-side orchestration ----------------
static inline size_t packed_halves(int Cout, int Cin, int KH, int KW) {
    int Coutp = (Cout + 15) & ~15;
    int nck   = (Cin + 31) / 32;
    return (size_t)KH * KW * nck * Coutp * 32;
}

extern "C" void kernel_launch(void* const* d_in, const int* in_sizes, int n_in,
                              void* d_out, int out_size, void* d_ws, size_t ws_size,
                              hipStream_t stream) {
    (void)in_sizes; (void)n_in; (void)out_size; (void)ws_size;

    const float* net0 = (const float*)d_in[0];
    const float* inp  = (const float*)d_in[1];
    const float* corr = (const float*)d_in[2];
    const float* flow = (const float*)d_in[3];

    const float* convc1_w = (const float*)d_in[4];
    const float* convc1_b = (const float*)d_in[5];
    const float* convc2_w = (const float*)d_in[6];
    const float* convc2_b = (const float*)d_in[7];
    const float* convf1_w = (const float*)d_in[8];
    const float* convf1_b = (const float*)d_in[9];
    const float* convf2_w = (const float*)d_in[10];
    const float* convf2_b = (const float*)d_in[11];
    const float* enc_w    = (const float*)d_in[12];
    const float* enc_b    = (const float*)d_in[13];
    const float* convz_w  = (const float*)d_in[14];
    const float* convz_b  = (const float*)d_in[15];
    const float* convr_w  = (const float*)d_in[16];
    const float* convr_b  = (const float*)d_in[17];
    const float* convq_w  = (const float*)d_in[18];
    const float* convq_b  = (const float*)d_in[19];
    const float* fh1_w    = (const float*)d_in[20];
    const float* fh2_w    = (const float*)d_in[21];
    const float* fh2_b    = (const float*)d_in[22];
    const float* mask1_w  = (const float*)d_in[23];
    const float* mask1_b  = (const float*)d_in[24];
    const float* mask2_w  = (const float*)d_in[25];
    const float* mask2_b  = (const float*)d_in[26];

    char*  ws  = (char*)d_ws;
    size_t off = 0;
    auto alloc = [&](size_t bytes) -> char* {
        char* p = ws + off;
        off = (off + bytes + 255) & ~(size_t)255;
        return p;
    };

    // packed weights (f16)
    _Float16* wp_c1 = (_Float16*)alloc(packed_halves(32, 36, 1, 1) * 2);
    _Float16* wp_c2 = (_Float16*)alloc(packed_halves(32, 32, 3, 3) * 2);
    _Float16* wp_f1 = (_Float16*)alloc(packed_halves(32, 1, 7, 7) * 2);
    _Float16* wp_f2 = (_Float16*)alloc(packed_halves(32, 32, 3, 3) * 2);
    _Float16* wp_en = (_Float16*)alloc(packed_halves(63, 64, 3, 3) * 2);
    _Float16* wp_z  = (_Float16*)alloc(packed_halves(128, 320, 3, 3) * 2);
    _Float16* wp_r  = (_Float16*)alloc(packed_halves(128, 320, 3, 3) * 2);
    _Float16* wp_q  = (_Float16*)alloc(packed_halves(128, 320, 3, 3) * 2);
    _Float16* wp_h1 = (_Float16*)alloc(packed_halves(128, 128, 3, 3) * 2);
    _Float16* wp_h2 = (_Float16*)alloc(packed_halves(1, 128, 1, 1) * 2);
    _Float16* wp_m1 = (_Float16*)alloc(packed_halves(128, 128, 3, 3) * 2);
    _Float16* wp_m2 = (_Float16*)alloc(packed_halves(144, 128, 1, 1) * 2);

    // NHWC f16 buffers (per-pixel channel counts, padded)
    const size_t px  = (size_t)NB_ * HW_;
    _Float16* zpad    = (_Float16*)alloc(256);
    _Float16* net0h   = (_Float16*)alloc(px * 128 * 2);
    _Float16* inph    = (_Float16*)alloc(px * 128 * 2);
    _Float16* corrh   = (_Float16*)alloc(px * 64 * 2);    // 36 -> pad 64
    _Float16* flowh   = (_Float16*)alloc(px * 32 * 2);    // 1  -> pad 32
    _Float16* cor1h   = (_Float16*)alloc(px * 32 * 2);
    _Float16* cath    = (_Float16*)alloc(px * 64 * 2);    // [cor2 | flo2]
    _Float16* flo1h   = (_Float16*)alloc(px * 32 * 2);
    _Float16* motionh = (_Float16*)alloc(px * 64 * 2);    // enc(63) + flow(1)
    _Float16* zh      = (_Float16*)alloc(px * 128 * 2);
    _Float16* rh      = (_Float16*)alloc(px * 128 * 2);   // r, later q
    _Float16* rnh     = (_Float16*)alloc(px * 128 * 2);   // r * net0
    _Float16* hh      = (_Float16*)alloc(px * 128 * 2);   // h (f16 copy)
    _Float16* dh      = (_Float16*)alloc(px * 128 * 2);   // fh1 out, then mask1 out

    auto launch_pack = [&](const float* w, _Float16* wp, int Cout, int Cin,
                           int KH, int KW) {
        size_t tot = packed_halves(Cout, Cin, KH, KW);
        int blocks = (int)((tot + 255) / 256);
        if (blocks > 4096) blocks = 4096;
        pack_weights<<<blocks, 256, 0, stream>>>(w, wp, Cout, Cin, KH, KW);
    };

    auto launch_conv = [&](const _Float16* a0, int C0p, const _Float16* a1, int C1p,
                           const _Float16* a2, int C2p, const _Float16* wp,
                           const float* bias,
                           _Float16* oh, int ohcs, int ohcoff, float* of, int ofcs,
                           int Cout, int KH, int KW, int pad, int dil,
                           int act, float scale) {
        int Coutp = (Cout + 15) & ~15;
        int gy    = ((Coutp >> 4) + 7) / 8;
        dim3 grid(NPTX, gy);
        conv_wmma<<<grid, 256, 0, stream>>>(a0, C0p, a1, C1p, a2, C2p, wp, bias,
                                            zpad, oh, ohcs, ohcoff, of, ofcs,
                                            Cout, KH, KW, pad, dil, act, scale);
    };

    // ---- weight packing + input layout conversion ----
    zero_h<<<1, 256, 0, stream>>>(zpad, 128);
    launch_pack(convc1_w, wp_c1, 32, 36, 1, 1);
    launch_pack(convc2_w, wp_c2, 32, 32, 3, 3);
    launch_pack(convf1_w, wp_f1, 32, 1, 7, 7);
    launch_pack(convf2_w, wp_f2, 32, 32, 3, 3);
    launch_pack(enc_w,    wp_en, 63, 64, 3, 3);
    launch_pack(convz_w,  wp_z, 128, 320, 3, 3);
    launch_pack(convr_w,  wp_r, 128, 320, 3, 3);
    launch_pack(convq_w,  wp_q, 128, 320, 3, 3);
    launch_pack(fh1_w,    wp_h1, 128, 128, 3, 3);
    launch_pack(fh2_w,    wp_h2, 1, 128, 1, 1);
    launch_pack(mask1_w,  wp_m1, 128, 128, 3, 3);
    launch_pack(mask2_w,  wp_m2, 144, 128, 1, 1);
    to_nhwc_h<<<4096, 256, 0, stream>>>(net0, net0h, 128, 128);
    to_nhwc_h<<<4096, 256, 0, stream>>>(inp,  inph,  128, 128);
    to_nhwc_h<<<4096, 256, 0, stream>>>(corr, corrh, 36, 64);
    to_nhwc_h<<<4096, 256, 0, stream>>>(flow, flowh, 1, 32);

    // ---- BasicMotionEncoder ----
    launch_conv(corrh, 64, nullptr, 0, nullptr, 0, wp_c1, convc1_b,
                cor1h, 32, 0, nullptr, 0, 32, 1, 1, 0, 1, ACT_RELU, 1.f);
    launch_conv(cor1h, 32, nullptr, 0, nullptr, 0, wp_c2, convc2_b,
                cath, 64, 0, nullptr, 0, 32, 3, 3, 1, 1, ACT_RELU, 1.f);
    launch_conv(flowh, 32, nullptr, 0, nullptr, 0, wp_f1, convf1_b,
                flo1h, 32, 0, nullptr, 0, 32, 7, 7, 3, 1, ACT_RELU, 1.f);
    launch_conv(flo1h, 32, nullptr, 0, nullptr, 0, wp_f2, convf2_b,
                cath, 64, 32, nullptr, 0, 32, 3, 3, 1, 1, ACT_RELU, 1.f);
    launch_conv(cath, 64, nullptr, 0, nullptr, 0, wp_en, enc_b,
                motionh, 64, 0, nullptr, 0, 63, 3, 3, 1, 1, ACT_RELU, 1.f);
    flow_to_motion<<<(int)((px + 255) / 256), 256, 0, stream>>>(flow, motionh);

    // ---- ConvGRU (3x3, pad=2, dil=2), hx = [net0|motion|inp] ----
    launch_conv(net0h, 128, motionh, 64, inph, 128, wp_z, convz_b,
                zh, 128, 0, nullptr, 0, 128, 3, 3, 2, 2, ACT_SIGM, 1.f);
    launch_conv(net0h, 128, motionh, 64, inph, 128, wp_r, convr_b,
                rh, 128, 0, nullptr, 0, 128, 3, 3, 2, 2, ACT_SIGM, 1.f);
    ew_mul_h<<<4096, 256, 0, stream>>>(rh, net0h, rnh, px * 128);
    launch_conv(rnh, 128, motionh, 64, inph, 128, wp_q, convq_b,
                rh /*q*/, 128, 0, nullptr, 0, 128, 3, 3, 2, 2, ACT_TANH, 1.f);

    float* hout  = (float*)d_out;                    // h: 8*128*HW f32 NCHW
    float* delta = hout + (size_t)NB_ * 128 * HW_;   // 8*1*HW
    float* mask  = hout + (size_t)NB_ * 129 * HW_;   // 8*144*HW
    gru_combine<<<4096, 256, 0, stream>>>(zh, rh, net0h, hout, hh);

    // ---- FlowHead ----
    launch_conv(hh, 128, nullptr, 0, nullptr, 0, wp_h1, nullptr,
                dh, 128, 0, nullptr, 0, 128, 3, 3, 2, 2, ACT_RELU, 1.f);
    launch_conv(dh, 128, nullptr, 0, nullptr, 0, wp_h2, fh2_b,
                nullptr, 0, 0, delta, 1, 1, 1, 1, 0, 1, ACT_NONE, 1.f);

    // ---- Upsample mask head ----
    launch_conv(hh, 128, nullptr, 0, nullptr, 0, wp_m1, mask1_b,
                dh /*m*/, 128, 0, nullptr, 0, 128, 3, 3, 1, 1, ACT_RELU, 1.f);
    launch_conv(dh, 128, nullptr, 0, nullptr, 0, wp_m2, mask2_b,
                nullptr, 0, 0, mask, 144, 144, 1, 1, 0, 1, ACT_NONE, 0.25f);
}
